// Mamba2InformedElman_49314814493168
// MI455X (gfx1250) — matrix operand order repaired and checked by reading, not compiled, not run on hardware
//
#include <hip/hip_runtime.h>
#include <hip/hip_bf16.h>

typedef __attribute__((ext_vector_type(16))) __bf16 v16bf;
typedef __attribute__((ext_vector_type(2)))  __bf16 v2bf;
typedef __attribute__((ext_vector_type(8)))  float  v8f;

#define DIM      1024
#define D_INNER  2048
#define NHEADS   16
#define HEADDIM  128
#define D_STATE  64
#define D_PROJ   4240   // 2*D_INNER + 2*D_STATE + NHEADS
#define BATCH    8
#define SEQ      1024
#define MTOT     (BATCH*SEQ)   // 8192

union FragAB { v16bf v; uint4 q[2]; };

// Native bf16 conversion (v_cvt_pk_bf16_f32 / v_cvt_bf16_f32, RNE).
__device__ __forceinline__ unsigned pk_bf16(float a, float b) {
  union { v2bf v; unsigned u; } t;
  t.v.x = (__bf16)a;
  t.v.y = (__bf16)b;
  return t.u;
}

__device__ __forceinline__ unsigned short f2bf(float a) {
  union { __bf16 b; unsigned short s; } t;
  t.b = (__bf16)a;
  return t.s;
}

// Fast sigmoid: native v_exp_f32 + native v_rcp_f32 (no precise-div sequence).
__device__ __forceinline__ float sigmoidf_(float x) {
  return __builtin_amdgcn_rcpf(1.0f + __expf(-x));
}

// ---------------------------------------------------------------------------
// Kernel A: proj = X @ W_in^T, fused activations.
//   X    [8192][1024] f32, W_in [4240][1024] f32
//   proj [8192][4240] f32: cols [0,2048)=silu(xp), [2048,4224)=z/B/C raw,
//        [4224,4240)=sigmoid(dt+dt_bias) (decay)
// 128x128x32 tiles, bf16 WMMA, f32 accumulate.
// 2-stage pipeline: global->reg prefetch + LDS double buffer, 1 barrier/iter.
// ---------------------------------------------------------------------------
__global__ __launch_bounds__(256)
void gemm_proj_kernel(const float* __restrict__ X,
                      const float* __restrict__ Win,
                      const float* __restrict__ dtb,
                      float* __restrict__ proj)
{
  __shared__ unsigned lA[2][128][20];  // [stage][m][kpair]; 16 kpairs=32 bf16 K
  __shared__ unsigned lB[2][128][20];  // [stage][n][kpair]

  const int tid   = threadIdx.x;
  const int mBase = blockIdx.y * 128;
  const int nBase = blockIdx.x * 128;
  const int wave  = tid >> 5, lane = tid & 31;
  const int wr = wave >> 1, wc = wave & 1;      // 4x2 wave grid: 32x64 each
  const int lg = lane >> 4, ln = lane & 15;

  v8f acc[2][4] = {};

  const int lrow = tid >> 1;          // 0..127
  const int koff = (tid & 1) * 16;    // 0 or 16 (floats along K)
  const float* xp = X + (size_t)(mBase + lrow) * DIM + koff;
  int wrow = nBase + lrow; if (wrow > D_PROJ - 1) wrow = D_PROJ - 1;  // clamp (discarded later)
  const float* wp = Win + (size_t)wrow * DIM + koff;

  float4 ra[4], rb[4];
  auto gload = [&]() {
    ra[0] = *(const float4*)(xp + 0);
    ra[1] = *(const float4*)(xp + 4);
    ra[2] = *(const float4*)(xp + 8);
    ra[3] = *(const float4*)(xp + 12);
    rb[0] = *(const float4*)(wp + 0);
    rb[1] = *(const float4*)(wp + 4);
    rb[2] = *(const float4*)(wp + 8);
    rb[3] = *(const float4*)(wp + 12);
  };
  auto lstore = [&](int s) {
    unsigned* pa = &lA[s][lrow][koff >> 1];
    pa[0] = pk_bf16(ra[0].x, ra[0].y); pa[1] = pk_bf16(ra[0].z, ra[0].w);
    pa[2] = pk_bf16(ra[1].x, ra[1].y); pa[3] = pk_bf16(ra[1].z, ra[1].w);
    pa[4] = pk_bf16(ra[2].x, ra[2].y); pa[5] = pk_bf16(ra[2].z, ra[2].w);
    pa[6] = pk_bf16(ra[3].x, ra[3].y); pa[7] = pk_bf16(ra[3].z, ra[3].w);
    unsigned* pb = &lB[s][lrow][koff >> 1];
    pb[0] = pk_bf16(rb[0].x, rb[0].y); pb[1] = pk_bf16(rb[0].z, rb[0].w);
    pb[2] = pk_bf16(rb[1].x, rb[1].y); pb[3] = pk_bf16(rb[1].z, rb[1].w);
    pb[4] = pk_bf16(rb[2].x, rb[2].y); pb[5] = pk_bf16(rb[2].z, rb[2].w);
    pb[6] = pk_bf16(rb[3].x, rb[3].y); pb[7] = pk_bf16(rb[3].z, rb[3].w);
  };

  gload();       // K-slice 0
  lstore(0);
  const int NIT = DIM / 32;
  for (int i = 0; i < NIT; ++i) {
    __syncthreads();
    const int cur = i & 1;
    const bool more = (i + 1 < NIT);
    if (more) { xp += 32; wp += 32; gload(); }   // prefetch next slice (hidden)

    FragAB afr[2], bfr[4];
    #pragma unroll
    for (int rr = 0; rr < 2; ++rr) {            // A 16x32 lane layout
      int m = wr*32 + rr*16 + ln;
      afr[rr].q[0] = *(const uint4*)&lA[cur][m][8*lg];
      afr[rr].q[1] = *(const uint4*)&lA[cur][m][8*lg + 4];
    }
    #pragma unroll
    for (int cc = 0; cc < 4; ++cc) {            // B 32x16 lane layout
      int n = wc*64 + cc*16 + ln;
      bfr[cc].q[0] = *(const uint4*)&lB[cur][n][8*lg];
      bfr[cc].q[1] = *(const uint4*)&lB[cur][n][8*lg + 4];
    }
    #pragma unroll
    for (int rr = 0; rr < 2; ++rr)
      #pragma unroll
      for (int cc = 0; cc < 4; ++cc)
        acc[rr][cc] = __builtin_amdgcn_wmma_f32_16x16x32_bf16(
            false, afr[rr].v, false, bfr[cc].v, (short)0, acc[rr][cc], false, false);

    if (more) lstore(cur ^ 1);                   // fill other stage for next iter
  }

  // Epilogue. Class boundaries (2048/4096/4224/4240) are multiples of 16, so
  // each 16-wide cc tile is class-uniform -> hoist class/bias out of r loops.
  #pragma unroll
  for (int cc = 0; cc < 4; ++cc) {
    const int gn = nBase + wc*64 + cc*16 + ln;
    const bool valid = gn < D_PROJ;
    int cls;
    if (gn < D_INNER) cls = 0;                         // silu(x_p)
    else if (gn < 2*D_INNER + 2*D_STATE) cls = 1;      // z / B / C raw
    else cls = 2;                                      // decay
    float bias = 0.0f;
    if (cls == 2 && valid) bias = dtb[gn - (2*D_INNER + 2*D_STATE)];
    #pragma unroll
    for (int rr = 0; rr < 2; ++rr)
      #pragma unroll
      for (int r = 0; r < 8; ++r) {
        float v = acc[rr][cc][r];
        if (cls == 0)      v = v * sigmoidf_(v);
        else if (cls == 2) v = sigmoidf_(v + bias);
        const int gm = mBase + wr*32 + rr*16 + lg*8 + r;
        if (valid) proj[(size_t)gm * D_PROJ + gn] = v;
      }
  }
}

// ---------------------------------------------------------------------------
// Kernel B: sequential scan over T, one workgroup per (b, h).
// State H[128][64] in registers: thread (p = tid>>1, half = tid&1) holds 32
// floats. B_t/C_t double-buffered in LDS (vectorized staging), 1 barrier/step.
// ---------------------------------------------------------------------------
__global__ __launch_bounds__(256)
void scan_kernel(const float* __restrict__ proj,
                 const float* __restrict__ H0,
                 unsigned short* __restrict__ ys,   // [8192][2048] bf16
                 float* __restrict__ Hout)          // [B][h][p][n] f32
{
  const int bh = blockIdx.x;           // 0..127
  const int b = bh >> 4, h = bh & 15;
  const int tid = threadIdx.x;
  const int p  = tid >> 1;
  const int nh = tid & 1;
  const int nbase = nh * 32;

  __shared__ __align__(16) float sB[2][64];
  __shared__ __align__(16) float sC[2][64];
  __shared__ float sDec[2];

  float hreg[32];
  {
    const float* h0p = H0 + ((size_t)(b*NHEADS + h)*HEADDIM + p)*D_STATE + nbase;
    #pragma unroll
    for (int j = 0; j < 32; ++j) hreg[j] = h0p[j];
  }

  // Vectorized staging: 16 threads for B, 16 for C (float4 each), 1 for decay.
  auto stage = [&](int t, int s) {
    size_t row = (size_t)(b*SEQ + t) * D_PROJ;
    if (tid < 16) {
      *(float4*)&sB[s][tid*4] = *(const float4*)(proj + row + 2*D_INNER + tid*4);
    } else if (tid < 32) {
      int i = tid - 16;
      *(float4*)&sC[s][i*4] = *(const float4*)(proj + row + 2*D_INNER + D_STATE + i*4);
    } else if (tid == 32) {
      sDec[s] = proj[row + 2*D_INNER + 2*D_STATE + h];
    }
  };

  stage(0, 0);
  int buf = 0;
  for (int t = 0; t < SEQ; ++t) {
    __syncthreads();
    if (t + 1 < SEQ) {   // stage next step's B/C/decay into the other buffer
      stage(t + 1, buf ^ 1);
      size_t rown = (size_t)(b*SEQ + t + 1) * D_PROJ;
      __builtin_prefetch(proj + rown + h*HEADDIM + p, 0, 0);
      __builtin_prefetch(proj + rown + D_INNER + h*HEADDIM + p, 0, 0);
    }
    size_t row = (size_t)(b*SEQ + t) * D_PROJ;
    float xv  = proj[row + h*HEADDIM + p];              // silu already applied
    float zv  = proj[row + D_INNER + h*HEADDIM + p];
    float dec = sDec[buf];

    float acc = 0.0f;
    #pragma unroll
    for (int j = 0; j < 32; ++j) {
      hreg[j] = fmaf(dec, hreg[j], xv * sB[buf][nbase + j]);
      acc = fmaf(hreg[j], sC[buf][nbase + j], acc);
    }
    acc += __shfl_xor(acc, 1, 32);    // combine the two n-halves (adjacent lanes)

    if (nh == 0) {
      float y = acc;
      float a = zv + y;
      float o = y * (a * sigmoidf_(a));                 // y * silu(z + y)
      ys[(size_t)(b*SEQ + t)*D_INNER + h*HEADDIM + p] = f2bf(o);
    }
    buf ^= 1;
  }

  float* hp = Hout + ((size_t)(b*NHEADS + h)*HEADDIM + p)*D_STATE + nbase;
  #pragma unroll
  for (int j = 0; j < 32; ++j) hp[j] = hreg[j];
}

// ---------------------------------------------------------------------------
// Kernel C: out = ys @ W_out^T.  ys [8192][2048] bf16, W_out [1024][2048] f32.
// Same pipelined 128x128x32 bf16-WMMA structure; A is already bf16.
// ---------------------------------------------------------------------------
__global__ __launch_bounds__(256)
void gemm_out_kernel(const unsigned* __restrict__ YS,   // [8192][1024] u32 (2x bf16)
                     const float* __restrict__ Wout,
                     float* __restrict__ out)           // [8192][1024] f32
{
  __shared__ unsigned lA[2][128][20];
  __shared__ unsigned lB[2][128][20];

  const int tid   = threadIdx.x;
  const int mBase = blockIdx.y * 128;
  const int nBase = blockIdx.x * 128;
  const int wave  = tid >> 5, lane = tid & 31;
  const int wr = wave >> 1, wc = wave & 1;
  const int lg = lane >> 4, ln = lane & 15;

  v8f acc[2][4] = {};

  const int lrow = tid >> 1;
  const int koff = (tid & 1) * 16;
  const unsigned* ap = YS + (size_t)(mBase + lrow) * (D_INNER/2) + (koff >> 1);
  const float*    wp = Wout + (size_t)(nBase + lrow) * D_INNER + koff;

  uint4  qa[2];
  float4 rb[4];
  auto gload = [&]() {
    qa[0] = *(const uint4*)(ap + 0);
    qa[1] = *(const uint4*)(ap + 4);
    rb[0] = *(const float4*)(wp + 0);
    rb[1] = *(const float4*)(wp + 4);
    rb[2] = *(const float4*)(wp + 8);
    rb[3] = *(const float4*)(wp + 12);
  };
  auto lstore = [&](int s) {
    *(uint4*)&lA[s][lrow][(koff >> 1) + 0] = qa[0];
    *(uint4*)&lA[s][lrow][(koff >> 1) + 4] = qa[1];
    unsigned* pb = &lB[s][lrow][koff >> 1];
    pb[0] = pk_bf16(rb[0].x, rb[0].y); pb[1] = pk_bf16(rb[0].z, rb[0].w);
    pb[2] = pk_bf16(rb[1].x, rb[1].y); pb[3] = pk_bf16(rb[1].z, rb[1].w);
    pb[4] = pk_bf16(rb[2].x, rb[2].y); pb[5] = pk_bf16(rb[2].z, rb[2].w);
    pb[6] = pk_bf16(rb[3].x, rb[3].y); pb[7] = pk_bf16(rb[3].z, rb[3].w);
  };

  gload();
  lstore(0);
  const int NIT = D_INNER / 32;
  for (int i = 0; i < NIT; ++i) {
    __syncthreads();
    const int cur = i & 1;
    const bool more = (i + 1 < NIT);
    if (more) { ap += 16; wp += 32; gload(); }

    FragAB afr[2], bfr[4];
    #pragma unroll
    for (int rr = 0; rr < 2; ++rr) {
      int m = wr*32 + rr*16 + ln;
      afr[rr].q[0] = *(const uint4*)&lA[cur][m][8*lg];
      afr[rr].q[1] = *(const uint4*)&lA[cur][m][8*lg + 4];
    }
    #pragma unroll
    for (int cc = 0; cc < 4; ++cc) {
      int n = wc*64 + cc*16 + ln;
      bfr[cc].q[0] = *(const uint4*)&lB[cur][n][8*lg];
      bfr[cc].q[1] = *(const uint4*)&lB[cur][n][8*lg + 4];
    }
    #pragma unroll
    for (int rr = 0; rr < 2; ++rr)
      #pragma unroll
      for (int cc = 0; cc < 4; ++cc)
        acc[rr][cc] = __builtin_amdgcn_wmma_f32_16x16x32_bf16(
            false, afr[rr].v, false, bfr[cc].v, (short)0, acc[rr][cc], false, false);

    if (more) lstore(cur ^ 1);
  }

  #pragma unroll
  for (int rr = 0; rr < 2; ++rr)
  #pragma unroll
  for (int cc = 0; cc < 4; ++cc)
  #pragma unroll
  for (int r = 0; r < 8; ++r) {
    const int gm = mBase + wr*32 + rr*16 + lg*8 + r;
    const int gn = nBase + wc*64 + cc*16 + ln;
    out[(size_t)gm * DIM + gn] = acc[rr][cc][r];
  }
}

// ---------------------------------------------------------------------------
extern "C" void kernel_launch(void* const* d_in, const int* in_sizes, int n_in,
                              void* d_out, int out_size, void* d_ws, size_t ws_size,
                              hipStream_t stream)
{
  (void)in_sizes; (void)n_in; (void)out_size; (void)ws_size;
  const float* x    = (const float*)d_in[0];
  const float* H0   = (const float*)d_in[1];
  const float* Win  = (const float*)d_in[2];
  const float* Wout = (const float*)d_in[3];
  const float* dtb  = (const float*)d_in[4];

  float* out  = (float*)d_out;
  float* Hout = out + (size_t)BATCH * SEQ * DIM;          // H_final after output

  float* proj = (float*)d_ws;                              // 8192*4240 f32
  size_t projBytes = (size_t)MTOT * D_PROJ * sizeof(float);
  unsigned short* ys = (unsigned short*)((char*)d_ws + projBytes);  // 8192*2048 bf16

  dim3 g1((D_PROJ + 127) / 128, MTOT / 128);   // 34 x 64
  gemm_proj_kernel<<<g1, 256, 0, stream>>>(x, Win, dtb, proj);

  scan_kernel<<<BATCH * NHEADS, 256, 0, stream>>>(proj, H0, ys, Hout);

  dim3 g3(DIM / 128, MTOT / 128);              // 8 x 64
  gemm_out_kernel<<<g3, 256, 0, stream>>>((const unsigned*)ys, Wout, out);
}
